// Seq2SeqAttention_90108413870861
// MI455X (gfx1250) — matrix operand order repaired and checked by reading, not compiled, not run on hardware
//
#include <hip/hip_runtime.h>
#include <hip/hip_bf16.h>

// ---------------------------------------------------------------------------
// MI455X (gfx1250) persistent seq2seq-attention implementation, round 3.
//  - All gate GEMMs via v_wmma_f32_16x16x32_bf16 (wave32, 16x16 tiles).
//  - Recurrent activations (h, ctx, enc1) kept bf16-resident in memory:
//    A-fragments load as two 16B v8bf chunks with zero per-step conversions.
//  - Tiny K-dims (curr/x with 2 features) handled on the VALU side-path so the
//    WMMA K-loop is always a clean multiple of 32 with chunk-aligned sources.
// ---------------------------------------------------------------------------

typedef __bf16 bf16_t;
typedef bf16_t v16bf __attribute__((ext_vector_type(16)));
typedef bf16_t v8bf  __attribute__((ext_vector_type(8)));
typedef float  v8f   __attribute__((ext_vector_type(8)));

static __device__ __forceinline__ float sigm(float x) { return 1.0f / (1.0f + __expf(-x)); }

static __device__ __forceinline__ v8f v8zero() {
  v8f z;
#pragma unroll
  for (int i = 0; i < 8; ++i) z[i] = 0.0f;
  return z;
}

static __device__ __forceinline__ v8f wmma_bf16(v16bf a, v16bf b, v8f c) {
  return __builtin_amdgcn_wmma_f32_16x16x32_bf16(false, a, false, b, (short)0, c, false, false);
}

// A fragment from a bf16 row: two contiguous 8-bf16 (16B) runs, no converts.
static __device__ __forceinline__ v16bf make_afrag_bf(const bf16_t* __restrict__ p, int khalf) {
  v8bf lo = *(const v8bf*)(p + khalf * 16);
  v8bf hi = *(const v8bf*)(p + khalf * 16 + 8);
  v16bf a;
#pragma unroll
  for (int i = 0; i < 8; ++i) { a[i] = lo[i]; a[8 + i] = hi[i]; }
  return a;
}

// A fragment from an fp32 row (used only in the one-shot keys projection).
static __device__ __forceinline__ v16bf make_afrag_vec(const float* __restrict__ p, int khalf) {
  const float4 x0 = *(const float4*)(p + khalf * 16);
  const float4 x1 = *(const float4*)(p + khalf * 16 + 4);
  const float4 y0 = *(const float4*)(p + khalf * 16 + 8);
  const float4 y1 = *(const float4*)(p + khalf * 16 + 12);
  v16bf a;
  a[0] = (bf16_t)x0.x;  a[1] = (bf16_t)x0.y;  a[2] = (bf16_t)x0.z;  a[3] = (bf16_t)x0.w;
  a[4] = (bf16_t)x1.x;  a[5] = (bf16_t)x1.y;  a[6] = (bf16_t)x1.z;  a[7] = (bf16_t)x1.w;
  a[8] = (bf16_t)y0.x;  a[9] = (bf16_t)y0.y;  a[10] = (bf16_t)y0.z; a[11] = (bf16_t)y0.w;
  a[12] = (bf16_t)y1.x; a[13] = (bf16_t)y1.y; a[14] = (bf16_t)y1.z; a[15] = (bf16_t)y1.w;
  return a;
}

// Device-wide barrier: sync[0]=arrive counter, sync[1]=generation.
static __device__ __forceinline__ void grid_barrier(int* sync, int nblk) {
  __syncthreads();
  if (threadIdx.x == 0) {
    __threadfence();
    int g = __hip_atomic_load(&sync[1], __ATOMIC_ACQUIRE, __HIP_MEMORY_SCOPE_AGENT);
    int prev = __hip_atomic_fetch_add(&sync[0], 1, __ATOMIC_ACQ_REL, __HIP_MEMORY_SCOPE_AGENT);
    if (prev == nblk - 1) {
      __hip_atomic_store(&sync[0], 0, __ATOMIC_RELAXED, __HIP_MEMORY_SCOPE_AGENT);
      __hip_atomic_fetch_add(&sync[1], 1, __ATOMIC_ACQ_REL, __HIP_MEMORY_SCOPE_AGENT);
    } else {
      while (__hip_atomic_load(&sync[1], __ATOMIC_ACQUIRE, __HIP_MEMORY_SCOPE_AGENT) == g)
        __builtin_amdgcn_s_sleep(1);
    }
  }
  __syncthreads();
}

__global__ void reset_sync_kernel(int* s) {
  if (threadIdx.x < 8) s[threadIdx.x] = 0;
}

// ---------------------------------------------------------------------------
// Conv 3x3 stride-2 pad-1 + ReLU (NCHW / OIHW)
// ---------------------------------------------------------------------------
__global__ void conv3x3_s2_relu(const float* __restrict__ x, const float* __restrict__ w,
                                const float* __restrict__ bias, float* __restrict__ y,
                                int B, int Cin, int Hin, int Win, int Cout, int Hout, int Wout) {
  int idx = blockIdx.x * blockDim.x + threadIdx.x;
  int total = B * Cout * Hout * Wout;
  if (idx >= total) return;
  int ow = idx % Wout;
  int tt = idx / Wout;
  int oh = tt % Hout; tt /= Hout;
  int co = tt % Cout;
  int b  = tt / Cout;
  float s = bias[co];
  for (int ci = 0; ci < Cin; ++ci) {
    const float* xp = x + (size_t)(b * Cin + ci) * Hin * Win;
    const float* wp = w + (co * Cin + ci) * 9;
#pragma unroll
    for (int kh = 0; kh < 3; ++kh) {
      int ih = oh * 2 - 1 + kh;
      if (ih < 0 || ih >= Hin) continue;
#pragma unroll
      for (int kw = 0; kw < 3; ++kw) {
        int iw = ow * 2 - 1 + kw;
        if (iw < 0 || iw >= Win) continue;
        s += xp[ih * Win + iw] * wp[kh * 3 + kw];
      }
    }
  }
  y[idx] = fmaxf(s, 0.0f);
}

// mean over 8x8 spatial: x3 [128][128][64] -> means [128][128]
__global__ void meanpool_kernel(const float* __restrict__ x, float* __restrict__ means) {
  int idx = blockIdx.x * blockDim.x + threadIdx.x;
  if (idx >= 128 * 128) return;
  const float* p = x + (size_t)idx * 64;
  float s = 0.0f;
#pragma unroll
  for (int i = 0; i < 16; ++i) {
    float4 v = *(const float4*)(p + i * 4);
    s += v.x + v.y + v.z + v.w;
  }
  means[idx] = s * (1.0f / 64.0f);
}

// env_feat = means @ ef_w^T + ef_b  -> written into keys[:, 240, :]
__global__ void envfeat_kernel(const float* __restrict__ means, const float* __restrict__ ef_w,
                               const float* __restrict__ ef_b, float* __restrict__ keys) {
  int idx = blockIdx.x * blockDim.x + threadIdx.x;
  if (idx >= 128 * 256) return;
  int g = idx & 255;
  int b = idx >> 8;
  float s = ef_b[g];
  const float* m = means + b * 128;
  const float* w = ef_w + g * 128;
#pragma unroll
  for (int c = 0; c < 32; ++c) {
    float4 mv = *(const float4*)(m + c * 4);
    float4 wv = *(const float4*)(w + c * 4);
    s += wv.x * mv.x + wv.y * mv.y + wv.z * mv.z + wv.w * mv.w;
  }
  keys[((size_t)b * 241 + 240) * 256 + g] = s;
}

// ---------------------------------------------------------------------------
// Persistent LSTM layer: grid = 16 WGs x 256 threads.
//   WG wg owns hidden units [wg*16, wg*16+16); 4 N-tiles = gates i,f,g,o.
//   Wave w owns batch M-tile w. Cell state c lives in registers.
//   h is bf16-resident (double buffered); A-fragments are raw 16B loads.
//   SMALLX: Din==2 (fp32 x via VALU epilogue), WMMA K = 256 (h only),
//           sequence output y_bf (bf16, feeds next layer's WMMA).
//   else:   Din==256 bf16 x (prev layer out), WMMA K = 512 ([x | h]),
//           sequence output y_f32 (fp32 keys, feeds attention math).
// ---------------------------------------------------------------------------
template <bool SMALLX>
__global__ __launch_bounds__(256) void lstm_layer_kernel(
    const float* __restrict__ x_f32,   // SMALLX: [128][T][2]
    const bf16_t* __restrict__ x_bf,   // !SMALLX: [128][T][256] bf16
    const float* __restrict__ Wih,
    const float* __restrict__ Whh,
    const float* __restrict__ bih, const float* __restrict__ bhh,
    bf16_t* __restrict__ y_bf,         // SMALLX out: [128][y_stride][256] bf16
    float* __restrict__ y_f32,         // !SMALLX out: [128][y_stride][256] f32
    int y_stride,
    bf16_t* __restrict__ hbuf,         // [2][128][256] bf16
    float* __restrict__ h_out, float* __restrict__ c_out,
    int T, int* sync) {
  constexpr int KC = SMALLX ? 8 : 16;
  extern __shared__ char smem_raw[];
  bf16_t* sW = (bf16_t*)smem_raw;      // [4 ntiles][KC][32 lanes][16] bf16

  const int wg = blockIdx.x, tid = threadIdx.x;
  const int lane = tid & 31, wave = tid >> 5;
  const int khalf = lane >> 4, mrow = lane & 15;
  const int mtile = wave;

  // stage weights into LDS (B-fragment layout)
  for (int e = tid; e < 4 * KC * 512; e += 256) {
    int q = e & 15, l = (e >> 4) & 31, kc = (e >> 9) % KC, n = (e >> 9) / KC;
    int k = kc * 32 + (l >> 4) * 16 + q;
    int col = n * 256 + wg * 16 + (l & 15);
    float v;
    if (SMALLX) v = Whh[col * 256 + k];
    else        v = (k < 256) ? Wih[col * 256 + k] : Whh[col * 256 + (k - 256)];
    sW[e] = (bf16_t)v;
  }
  float bz[4], wx0[4], wx1[4];
#pragma unroll
  for (int n = 0; n < 4; ++n) {
    int col = n * 256 + wg * 16 + mrow;
    bz[n] = bih[col] + bhh[col];
    if (SMALLX) { wx0[n] = Wih[col * 2 + 0]; wx1[n] = Wih[col * 2 + 1]; }
    else        { wx0[n] = 0.0f; wx1[n] = 0.0f; }
  }
  float creg[8];
#pragma unroll
  for (int r = 0; r < 8; ++r) creg[r] = 0.0f;
  {  // zero h buffer 0 for owned cells
    int u = wg * 16 + mrow;
#pragma unroll
    for (int r = 0; r < 8; ++r) hbuf[(mtile * 16 + r + 8 * khalf) * 256 + u] = (bf16_t)0.0f;
  }
  grid_barrier(sync, 16);

  const int arow = mtile * 16 + mrow;
  const int ucol = wg * 16 + mrow;
  for (int t = 0; t < T; ++t) {
    const bf16_t* hc = hbuf + (t & 1) * (128 * 256);
    bf16_t* hn = hbuf + ((t + 1) & 1) * (128 * 256);
    v8f acc[4];
#pragma unroll
    for (int n = 0; n < 4; ++n) acc[n] = v8zero();

    if (!SMALLX) {  // x part: chunks 0..7 from bf16 x row
      const bf16_t* xrow = x_bf + ((size_t)arow * T + t) * 256;
#pragma unroll
      for (int kc = 0; kc < 8; ++kc) {
        v16bf a = make_afrag_bf(xrow + kc * 32, khalf);
#pragma unroll
        for (int n = 0; n < 4; ++n) {
          v16bf bfrag = *(const v16bf*)&sW[((n * KC + kc) * 32 + lane) * 16];
          acc[n] = wmma_bf16(a, bfrag, acc[n]);
        }
      }
    }
    {  // h part: chunks (SMALLX?0:8)..KC-1 from bf16 h buffer
      const bf16_t* hrow = hc + arow * 256;
#pragma unroll
      for (int kk = 0; kk < 8; ++kk) {
        const int kc = (SMALLX ? 0 : 8) + kk;
        v16bf a = make_afrag_bf(hrow + kk * 32, khalf);
#pragma unroll
        for (int n = 0; n < 4; ++n) {
          v16bf bfrag = *(const v16bf*)&sW[((n * KC + kc) * 32 + lane) * 16];
          acc[n] = wmma_bf16(a, bfrag, acc[n]);
        }
      }
    }
#pragma unroll
    for (int r = 0; r < 8; ++r) {
      int b = mtile * 16 + r + 8 * khalf;
      float g0 = 0.0f, g1 = 0.0f, g2 = 0.0f, g3 = 0.0f;
      if (SMALLX) {
        float2 xv = *(const float2*)&x_f32[((size_t)b * T + t) * 2];
        g0 = xv.x * wx0[0] + xv.y * wx1[0];
        g1 = xv.x * wx0[1] + xv.y * wx1[1];
        g2 = xv.x * wx0[2] + xv.y * wx1[2];
        g3 = xv.x * wx0[3] + xv.y * wx1[3];
      }
      float iv = sigm(acc[0][r] + g0 + bz[0]);
      float fv = sigm(acc[1][r] + g1 + bz[1]);
      float gv = tanhf(acc[2][r] + g2 + bz[2]);
      float ov = sigm(acc[3][r] + g3 + bz[3]);
      creg[r] = fv * creg[r] + iv * gv;
      float hv = ov * tanhf(creg[r]);
      hn[b * 256 + ucol] = (bf16_t)hv;
      if (SMALLX) y_bf[((size_t)b * y_stride + t) * 256 + ucol] = (bf16_t)hv;
      else        y_f32[((size_t)b * y_stride + t) * 256 + ucol] = hv;
      if (t == T - 1) {
        h_out[b * 256 + ucol] = hv;
        c_out[b * 256 + ucol] = creg[r];
      }
    }
    grid_barrier(sync, 16);
  }
}

// ---------------------------------------------------------------------------
// keys_proj = keys @ Wk^T + aW_b   (M = 128*241, N = 256, K = 256)
// Wk = aW_w[:, 256:512]. One-shot kernel; fp32 sources converted inline.
// ---------------------------------------------------------------------------
__global__ __launch_bounds__(256) void keysproj_kernel(const float* __restrict__ keys,
                                                       const float* __restrict__ aW_w,
                                                       const float* __restrict__ aW_b,
                                                       float* __restrict__ kp) {
  const int lane = threadIdx.x & 31, wave = threadIdx.x >> 5;
  const int khalf = lane >> 4, mrow = lane & 15;
  const int mtile = blockIdx.x;
  const int ntile = blockIdx.y * 8 + wave;
  const int row = mtile * 16 + mrow;
  const int col = ntile * 16 + mrow;
  const float* arow = keys + (size_t)row * 256;
  const float* wbase = aW_w + (size_t)col * 512 + 256;
  v8f acc = v8zero();
#pragma unroll
  for (int kc = 0; kc < 8; ++kc) {
    v16bf a = make_afrag_vec(arow + kc * 32, khalf);
    const float* wp = wbase + kc * 32 + khalf * 16;
    float4 w0 = *(const float4*)(wp + 0);
    float4 w1 = *(const float4*)(wp + 4);
    float4 w2 = *(const float4*)(wp + 8);
    float4 w3 = *(const float4*)(wp + 12);
    v16bf bfrag;
    bfrag[0] = (bf16_t)w0.x;  bfrag[1] = (bf16_t)w0.y;  bfrag[2] = (bf16_t)w0.z;  bfrag[3] = (bf16_t)w0.w;
    bfrag[4] = (bf16_t)w1.x;  bfrag[5] = (bf16_t)w1.y;  bfrag[6] = (bf16_t)w1.z;  bfrag[7] = (bf16_t)w1.w;
    bfrag[8] = (bf16_t)w2.x;  bfrag[9] = (bf16_t)w2.y;  bfrag[10] = (bf16_t)w2.z; bfrag[11] = (bf16_t)w2.w;
    bfrag[12] = (bf16_t)w3.x; bfrag[13] = (bf16_t)w3.y; bfrag[14] = (bf16_t)w3.z; bfrag[15] = (bf16_t)w3.w;
    acc = wmma_bf16(a, bfrag, acc);
  }
  float bias = aW_b[col];
#pragma unroll
  for (int r = 0; r < 8; ++r)
    kp[((size_t)(mtile * 16 + r + 8 * khalf)) * 256 + col] = acc[r] + bias;
}

// ---------------------------------------------------------------------------
// Persistent attention decoder: 360 steps, 5 grid-synced phases / step.
// hd1/hd2/ctx bf16-resident; qbuf/keys/keys_proj fp32 for the attention math.
// ---------------------------------------------------------------------------
__global__ __launch_bounds__(256) void decoder_kernel(
    const float* __restrict__ history,     // [128][240][2]
    const float* __restrict__ keys,        // [128][241][256]
    const float* __restrict__ keysp,       // [128][241][256]
    const float* __restrict__ aW_w,        // [256][512]
    const float* __restrict__ av_w, const float* __restrict__ av_b,
    const float* __restrict__ d0_Wih, const float* __restrict__ d0_Whh,
    const float* __restrict__ d0_bih, const float* __restrict__ d0_bhh,
    const float* __restrict__ d1_Wih, const float* __restrict__ d1_Whh,
    const float* __restrict__ d1_bih, const float* __restrict__ d1_bhh,
    const float* __restrict__ o_w, const float* __restrict__ o_b,
    const float* __restrict__ h1f, const float* __restrict__ c1f,
    const float* __restrict__ h2f, const float* __restrict__ c2f,
    bf16_t* __restrict__ hd1, bf16_t* __restrict__ hd2,     // [2][128][256] bf16
    float* __restrict__ qbuf,                               // [128][256] f32
    bf16_t* __restrict__ ctxbuf,                            // [128][256] bf16
    float* __restrict__ currbuf,                            // [2][128][2]
    float* __restrict__ preds,                              // [128][360][2]
    int* sync) {
  extern __shared__ char smem_raw[];
  bf16_t* sW0 = (bf16_t*)smem_raw;       // d0 [ctx|hd1]: 4*16*512
  bf16_t* sW1 = sW0 + 4 * 16 * 512;      // d1 [hd1|hd2]: 4*16*512
  bf16_t* sWq = sW1 + 4 * 16 * 512;      // Wq: 8*512
  float* sE = (float*)(smem_raw + (size_t)(4 * 16 * 512 + 4 * 16 * 512 + 8 * 512) * 2);
  float* sA = sE + 256;
  float* sR = sA + 256;

  const int wg = blockIdx.x, tid = threadIdx.x;
  const int lane = tid & 31, wave = tid >> 5;
  const int khalf = lane >> 4, mrow = lane & 15;
  const int mtile = wave;
  const int HD = 128 * 256;

  // --- stage constant weights once ---
  for (int e = tid; e < 4 * 16 * 512; e += 256) {   // d0: K = [ctx(256) | hd1(256)]
    int q = e & 15, l = (e >> 4) & 31, kc = (e >> 9) & 15, n = (e >> 9) >> 4;
    int k = kc * 32 + (l >> 4) * 16 + q;
    int col = n * 256 + wg * 16 + (l & 15);
    float v = (k < 256) ? d0_Wih[col * 258 + 2 + k] : d0_Whh[col * 256 + (k - 256)];
    sW0[e] = (bf16_t)v;
  }
  for (int e = tid; e < 4 * 16 * 512; e += 256) {   // d1: K = [hd1(256) | hd2(256)]
    int q = e & 15, l = (e >> 4) & 31, kc = (e >> 9) & 15, n = (e >> 9) >> 4;
    int k = kc * 32 + (l >> 4) * 16 + q;
    int col = n * 256 + wg * 16 + (l & 15);
    float v = (k < 256) ? d1_Wih[col * 256 + k] : d1_Whh[col * 256 + (k - 256)];
    sW1[e] = (bf16_t)v;
  }
  for (int e = tid; e < 8 * 512; e += 256) {        // Wq = aW_w[:, :256]
    int q = e & 15, l = (e >> 4) & 31, kc = (e >> 9);
    int k = kc * 32 + (l >> 4) * 16 + q;
    int col = wg * 16 + (l & 15);
    sWq[e] = (bf16_t)aW_w[col * 512 + k];
  }

  float b0[4], b1[4], pw0[4], pw1[4];
#pragma unroll
  for (int n = 0; n < 4; ++n) {
    int col = n * 256 + wg * 16 + mrow;
    b0[n] = d0_bih[col] + d0_bhh[col];
    b1[n] = d1_bih[col] + d1_bhh[col];
    pw0[n] = d0_Wih[col * 258 + 0];   // weights for curr.x / curr.y
    pw1[n] = d0_Wih[col * 258 + 1];
  }
  const int ucol = wg * 16 + mrow;
  float cd1r[8], cd2r[8];
#pragma unroll
  for (int r = 0; r < 8; ++r) {
    int b = mtile * 16 + r + 8 * khalf;
    cd1r[r] = c1f[b * 256 + ucol];
    cd2r[r] = c2f[b * 256 + ucol];
  }
  {  // init global state buffers (buffer 0); convert encoder finals to bf16
    int gid = wg * 256 + tid;
    for (int i = gid; i < HD; i += 16 * 256) {
      hd1[i] = (bf16_t)h1f[i];
      hd2[i] = (bf16_t)h2f[i];
    }
    if (gid < 256) {
      int b = gid >> 1, j = gid & 1;
      currbuf[gid] = history[(b * 240 + 239) * 2 + j];
    }
  }
  grid_barrier(sync, 16);

  const int arow = mtile * 16 + mrow;
  int cur = 0;
  for (int t = 0; t < 360; ++t) {
    const bf16_t* hd1c = hd1 + cur * HD;
    bf16_t* hd1n = hd1 + (cur ^ 1) * HD;
    const bf16_t* hd2c = hd2 + cur * HD;
    bf16_t* hd2n = hd2 + (cur ^ 1) * HD;
    const float* currc = currbuf + cur * 256;
    float* currn = currbuf + (cur ^ 1) * 256;

    // ---- Phase A: q = hd2 @ Wq^T ----
    {
      const bf16_t* hrow = hd2c + arow * 256;
      v8f acc = v8zero();
#pragma unroll
      for (int kc = 0; kc < 8; ++kc) {
        v16bf a = make_afrag_bf(hrow + kc * 32, khalf);
        v16bf bfrag = *(const v16bf*)&sWq[(kc * 32 + lane) * 16];
        acc = wmma_bf16(a, bfrag, acc);
      }
#pragma unroll
      for (int r = 0; r < 8; ++r)
        qbuf[(mtile * 16 + r + 8 * khalf) * 256 + ucol] = acc[r];
    }
    grid_barrier(sync, 16);

    // ---- Phase B: attention softmax + context (8 batches per WG) ----
    for (int bi = 0; bi < 8; ++bi) {
      int b = wg * 8 + bi;
      float e = -3.0e38f;
      if (tid < 241) {
        const float* kpp = keysp + ((size_t)b * 241 + tid) * 256;
        const float* qb = qbuf + b * 256;
        float dot = av_b[0];
#pragma unroll 4
        for (int h = 0; h < 64; ++h) {
          float4 qv = *(const float4*)(qb + h * 4);
          float4 kv = *(const float4*)(kpp + h * 4);
          float4 av = *(const float4*)(av_w + h * 4);
          dot += av.x * tanhf(qv.x + kv.x);
          dot += av.y * tanhf(qv.y + kv.y);
          dot += av.z * tanhf(qv.z + kv.z);
          dot += av.w * tanhf(qv.w + kv.w);
        }
        e = dot;
      }
      sE[tid] = e;
      sR[tid] = e;
      __syncthreads();
      for (int st = 128; st > 0; st >>= 1) {
        if (tid < st) sR[tid] = fmaxf(sR[tid], sR[tid + st]);
        __syncthreads();
      }
      float mx = sR[0];
      __syncthreads();
      float p = (tid < 241) ? __expf(sE[tid] - mx) : 0.0f;
      sA[tid] = p;
      sR[tid] = p;
      __syncthreads();
      for (int st = 128; st > 0; st >>= 1) {
        if (tid < st) sR[tid] += sR[tid + st];
        __syncthreads();
      }
      float inv = 1.0f / sR[0];
      __syncthreads();
      float accx = 0.0f;
      const float* kb = keys + (size_t)b * 241 * 256 + tid;
      for (int k = 0; k < 241; ++k) accx += sA[k] * kb[k * 256];
      ctxbuf[b * 256 + tid] = (bf16_t)(accx * inv);
      __syncthreads();
    }
    grid_barrier(sync, 16);

    // ---- Phase C: cell d0, WMMA input [ctx | hd1], curr via VALU epilogue ----
    {
      v8f acc[4];
#pragma unroll
      for (int n = 0; n < 4; ++n) acc[n] = v8zero();
      const bf16_t* crow = ctxbuf + arow * 256;
#pragma unroll
      for (int kc = 0; kc < 8; ++kc) {
        v16bf a = make_afrag_bf(crow + kc * 32, khalf);
#pragma unroll
        for (int n = 0; n < 4; ++n) {
          v16bf bfrag = *(const v16bf*)&sW0[((n * 16 + kc) * 32 + lane) * 16];
          acc[n] = wmma_bf16(a, bfrag, acc[n]);
        }
      }
      const bf16_t* hrow = hd1c + arow * 256;
#pragma unroll
      for (int kk = 0; kk < 8; ++kk) {
        const int kc = 8 + kk;
        v16bf a = make_afrag_bf(hrow + kk * 32, khalf);
#pragma unroll
        for (int n = 0; n < 4; ++n) {
          v16bf bfrag = *(const v16bf*)&sW0[((n * 16 + kc) * 32 + lane) * 16];
          acc[n] = wmma_bf16(a, bfrag, acc[n]);
        }
      }
#pragma unroll
      for (int r = 0; r < 8; ++r) {
        int b = mtile * 16 + r + 8 * khalf;
        float2 cv = *(const float2*)&currc[b * 2];
        float iv = sigm(acc[0][r] + cv.x * pw0[0] + cv.y * pw1[0] + b0[0]);
        float fv = sigm(acc[1][r] + cv.x * pw0[1] + cv.y * pw1[1] + b0[1]);
        float gv = tanhf(acc[2][r] + cv.x * pw0[2] + cv.y * pw1[2] + b0[2]);
        float ov = sigm(acc[3][r] + cv.x * pw0[3] + cv.y * pw1[3] + b0[3]);
        cd1r[r] = fv * cd1r[r] + iv * gv;
        hd1n[b * 256 + ucol] = (bf16_t)(ov * tanhf(cd1r[r]));
      }
    }
    grid_barrier(sync, 16);

    // ---- Phase D: cell d1, WMMA input [hd1_new | hd2] ----
    {
      v8f acc[4];
#pragma unroll
      for (int n = 0; n < 4; ++n) acc[n] = v8zero();
      const bf16_t* h1row = hd1n + arow * 256;
#pragma unroll
      for (int kc = 0; kc < 8; ++kc) {
        v16bf a = make_afrag_bf(h1row + kc * 32, khalf);
#pragma unroll
        for (int n = 0; n < 4; ++n) {
          v16bf bfrag = *(const v16bf*)&sW1[((n * 16 + kc) * 32 + lane) * 16];
          acc[n] = wmma_bf16(a, bfrag, acc[n]);
        }
      }
      const bf16_t* h2row = hd2c + arow * 256;
#pragma unroll
      for (int kk = 0; kk < 8; ++kk) {
        const int kc = 8 + kk;
        v16bf a = make_afrag_bf(h2row + kk * 32, khalf);
#pragma unroll
        for (int n = 0; n < 4; ++n) {
          v16bf bfrag = *(const v16bf*)&sW1[((n * 16 + kc) * 32 + lane) * 16];
          acc[n] = wmma_bf16(a, bfrag, acc[n]);
        }
      }
#pragma unroll
      for (int r = 0; r < 8; ++r) {
        int b = mtile * 16 + r + 8 * khalf;
        float iv = sigm(acc[0][r] + b1[0]);
        float fv = sigm(acc[1][r] + b1[1]);
        float gv = tanhf(acc[2][r] + b1[2]);
        float ov = sigm(acc[3][r] + b1[3]);
        cd2r[r] = fv * cd2r[r] + iv * gv;
        hd2n[b * 256 + ucol] = (bf16_t)(ov * tanhf(cd2r[r]));
      }
    }
    grid_barrier(sync, 16);

    // ---- Phase E: curr += hd2 @ o_w^T + o_b ; emit prediction ----
    if (wg == 0) {
      int b = tid >> 1, j = tid & 1;
      float s = o_b[j];
      const bf16_t* hp = hd2n + b * 256;
      const float* wp = o_w + j * 256;
#pragma unroll
      for (int h = 0; h < 32; ++h) {
        v8bf hv = *(const v8bf*)(hp + h * 8);
#pragma unroll
        for (int u = 0; u < 8; ++u) s += (float)hv[u] * wp[h * 8 + u];
      }
      float nc = currc[b * 2 + j] + s;
      currn[b * 2 + j] = nc;
      preds[((size_t)b * 360 + t) * 2 + j] = nc;
    }
    grid_barrier(sync, 16);
    cur ^= 1;
  }
}

// ---------------------------------------------------------------------------
// Host launcher
// ---------------------------------------------------------------------------
extern "C" void kernel_launch(void* const* d_in, const int* in_sizes, int n_in,
                              void* d_out, int out_size, void* d_ws, size_t ws_size,
                              hipStream_t stream) {
  (void)in_sizes; (void)n_in; (void)out_size; (void)ws_size;
  const float* history = (const float*)d_in[0];
  const float* env_map = (const float*)d_in[1];
  const float* e0_Wih = (const float*)d_in[2];
  const float* e0_Whh = (const float*)d_in[3];
  const float* e0_bih = (const float*)d_in[4];
  const float* e0_bhh = (const float*)d_in[5];
  const float* e1_Wih = (const float*)d_in[6];
  const float* e1_Whh = (const float*)d_in[7];
  const float* e1_bih = (const float*)d_in[8];
  const float* e1_bhh = (const float*)d_in[9];
  const float* c1w = (const float*)d_in[10];
  const float* c1b = (const float*)d_in[11];
  const float* c2w = (const float*)d_in[12];
  const float* c2b = (const float*)d_in[13];
  const float* c3w = (const float*)d_in[14];
  const float* c3b = (const float*)d_in[15];
  const float* ef_w = (const float*)d_in[16];
  const float* ef_b = (const float*)d_in[17];
  const float* aW_w = (const float*)d_in[18];
  const float* aW_b = (const float*)d_in[19];
  const float* av_w = (const float*)d_in[20];
  const float* av_b = (const float*)d_in[21];
  const float* d0_Wih = (const float*)d_in[22];
  const float* d0_Whh = (const float*)d_in[23];
  const float* d0_bih = (const float*)d_in[24];
  const float* d0_bhh = (const float*)d_in[25];
  const float* d1_Wih = (const float*)d_in[26];
  const float* d1_Whh = (const float*)d_in[27];
  const float* d1_bih = (const float*)d_in[28];
  const float* d1_bhh = (const float*)d_in[29];
  const float* o_w = (const float*)d_in[30];
  const float* o_b = (const float*)d_in[31];

  char* ws = (char*)d_ws;
  size_t off = 0;
  auto take = [&](size_t bytes) -> char* {
    off = (off + 255) & ~(size_t)255;
    char* p = ws + off;
    off += bytes;
    return p;
  };
  int* sync    = (int*)take(256);
  float* x1    = (float*)take((size_t)128 * 32 * 32 * 32 * 4);
  float* x2    = (float*)take((size_t)128 * 64 * 16 * 16 * 4);
  float* x3    = (float*)take((size_t)128 * 128 * 8 * 8 * 4);
  float* means = (float*)take((size_t)128 * 128 * 4);
  bf16_t* enc1 = (bf16_t*)take((size_t)128 * 240 * 256 * 2);   // bf16
  float* keys  = (float*)take((size_t)128 * 241 * 256 * 4);
  float* keysp = (float*)take((size_t)128 * 241 * 256 * 4);
  bf16_t* henc = (bf16_t*)take((size_t)2 * 128 * 256 * 2);     // bf16
  float* h1f   = (float*)take((size_t)128 * 256 * 4);
  float* c1f   = (float*)take((size_t)128 * 256 * 4);
  float* h2f   = (float*)take((size_t)128 * 256 * 4);
  float* c2f   = (float*)take((size_t)128 * 256 * 4);
  bf16_t* hd1  = (bf16_t*)take((size_t)2 * 128 * 256 * 2);     // bf16
  bf16_t* hd2  = (bf16_t*)take((size_t)2 * 128 * 256 * 2);     // bf16
  float* qb    = (float*)take((size_t)128 * 256 * 4);
  bf16_t* ctx  = (bf16_t*)take((size_t)128 * 256 * 2);         // bf16
  float* curr  = (float*)take((size_t)2 * 128 * 2 * 4);

  const int LSTM0_SMEM = 4 * 8 * 512 * 2;    // 32 KB (K=256)
  const int LSTM1_SMEM = 4 * 16 * 512 * 2;   // 64 KB (K=512)
  const int DEC_SMEM   = (4 * 16 * 512 + 4 * 16 * 512 + 8 * 512) * 2 + 3 * 256 * 4;  // ~139 KB

  // Allow >64KB dynamic LDS (WGP has 320KB); idempotent, capture-safe host call.
  (void)hipFuncSetAttribute((const void*)lstm_layer_kernel<false>,
                            hipFuncAttributeMaxDynamicSharedMemorySize, LSTM1_SMEM);
  (void)hipFuncSetAttribute((const void*)decoder_kernel,
                            hipFuncAttributeMaxDynamicSharedMemorySize, DEC_SMEM);

  reset_sync_kernel<<<1, 32, 0, stream>>>(sync);

  // CNN stack
  conv3x3_s2_relu<<<16384, 256, 0, stream>>>(env_map, c1w, c1b, x1, 128, 18, 64, 64, 32, 32, 32);
  conv3x3_s2_relu<<<8192, 256, 0, stream>>>(x1, c2w, c2b, x2, 128, 32, 32, 32, 64, 16, 16);
  conv3x3_s2_relu<<<4096, 256, 0, stream>>>(x2, c3w, c3b, x3, 128, 64, 16, 16, 128, 8, 8);
  meanpool_kernel<<<64, 256, 0, stream>>>(x3, means);
  envfeat_kernel<<<128, 256, 0, stream>>>(means, ef_w, ef_b, keys);  // keys[:,240,:]

  // Encoder LSTMs (persistent, WMMA bf16, bf16-resident h)
  lstm_layer_kernel<true><<<16, 256, LSTM0_SMEM, stream>>>(
      history, nullptr, e0_Wih, e0_Whh, e0_bih, e0_bhh,
      enc1, nullptr, 240, henc, h1f, c1f, 240, sync);
  lstm_layer_kernel<false><<<16, 256, LSTM1_SMEM, stream>>>(
      nullptr, enc1, e1_Wih, e1_Whh, e1_bih, e1_bhh,
      nullptr, keys, 241, henc, h2f, c2f, 240, sync);

  // keys_proj = keys @ Wk^T + aW_b
  keysproj_kernel<<<dim3(1928, 2), 256, 0, stream>>>(keys, aW_w, aW_b, keysp);

  // Attention decoder (persistent)
  decoder_kernel<<<16, 256, DEC_SMEM, stream>>>(
      history, keys, keysp, aW_w, av_w, av_b,
      d0_Wih, d0_Whh, d0_bih, d0_bhh,
      d1_Wih, d1_Whh, d1_bih, d1_bhh,
      o_w, o_b, h1f, c1f, h2f, c2f,
      hd1, hd2, qb, ctx, curr, (float*)d_out, sync);
}
